// TransitionUp_73839077753156
// MI455X (gfx1250) — matrix operand order repaired and checked by reading, not compiled, not run on hardware
//
#include <hip/hip_runtime.h>
#include <stdint.h>

typedef float v2f __attribute__((ext_vector_type(2)));
typedef float v8f __attribute__((ext_vector_type(8)));
typedef _Float16 v16h __attribute__((ext_vector_type(16)));

#define BATCH 4
#define NPTS1 2048
#define NPTS2 8192
#define FDIM 256
#define M1 (BATCH * NPTS1)   // 8192
#define M2 (BATCH * NPTS2)   // 32768
#define BN_EPS 1e-5f

// ---------------------------------------------------------------------------
// H[M x 256] = X[M x 256] @ W[256 x 256] + bias   (one wave per 16x16 tile)
// ---------------------------------------------------------------------------
__global__ __launch_bounds__(128) void gemm_bias_wmma(
    const float* __restrict__ X, const float* __restrict__ W,
    const float* __restrict__ bias, float* __restrict__ H) {
  const int lane = threadIdx.x;       // 0..31 (wave32)
  const int half = lane >> 4;         // 0 or 1
  const int l15  = lane & 15;
  const int tile_n = blockIdx.x * 4 + threadIdx.y;  // 0..15
  const int tile_m = blockIdx.y;
  const int arow = tile_m * 16 + l15; // A-matrix row for this lane
  const int col  = tile_n * 16 + l15; // B/C/D column for this lane

  v8f c = {};

#if __has_builtin(__builtin_amdgcn_wmma_f32_16x16x4_f32)
  // Native f32 WMMA: A 16x4 (VGPR0:K+0 / lanes16-31:K+2; VGPR1:K+1 / K+3)
  #pragma unroll 4
  for (int k0 = 0; k0 < FDIM; k0 += 4) {
    const int kb = k0 + half * 2;
    v2f a = *(const v2f*)(X + arow * FDIM + kb);   // consecutive K, 8B aligned
    v2f b;
    b.x = W[kb * FDIM + col];
    b.y = W[(kb + 1) * FDIM + col];
    c = __builtin_amdgcn_wmma_f32_16x16x4_f32(false, a, false, b,
                                              (short)0, c, false, false);
  }
#else
  // Fallback: f32 -> f16 hi/lo split, 3 WMMAs per K=32 step (~22-bit mantissa)
  for (int k0 = 0; k0 < FDIM; k0 += 32) {
    v16h aHi, aLo, bHi, bLo;
    #pragma unroll
    for (int v = 0; v < 8; ++v) {
      const int kb = k0 + ((v >= 4) ? 16 : 0) + half * 8 + (v & 3) * 2;
      float x0 = X[arow * FDIM + kb];
      float x1 = X[arow * FDIM + kb + 1];
      _Float16 h0 = (_Float16)x0, h1 = (_Float16)x1;
      aHi[2 * v] = h0;                      aHi[2 * v + 1] = h1;
      aLo[2 * v] = (_Float16)(x0 - (float)h0);
      aLo[2 * v + 1] = (_Float16)(x1 - (float)h1);
      const int kr = k0 + ((v >= 4) ? 16 : 0) + half * 8 + (v & 3) * 2;
      float w0 = W[kr * FDIM + col];
      float w1 = W[(kr + 1) * FDIM + col];
      _Float16 g0 = (_Float16)w0, g1 = (_Float16)w1;
      bHi[2 * v] = g0;                      bHi[2 * v + 1] = g1;
      bLo[2 * v] = (_Float16)(w0 - (float)g0);
      bLo[2 * v + 1] = (_Float16)(w1 - (float)g1);
    }
    c = __builtin_amdgcn_wmma_f32_16x16x32_f16(false, aHi, false, bHi, (short)0, c, false, false);
    c = __builtin_amdgcn_wmma_f32_16x16x32_f16(false, aLo, false, bHi, (short)0, c, false, false);
    c = __builtin_amdgcn_wmma_f32_16x16x32_f16(false, aHi, false, bLo, (short)0, c, false, false);
  }
#endif

  const float bv = bias[col];
  #pragma unroll
  for (int r = 0; r < 8; ++r) {
    const int orow = tile_m * 16 + half * 8 + r;  // C/D layout: lanes16-31 = M+8
    H[orow * FDIM + col] = c[r] + bv;
  }
}

// ---------------------------------------------------------------------------
// Per-feature batch-norm stats folded into affine: scale, shift
// ---------------------------------------------------------------------------
__global__ __launch_bounds__(256) void bn_stats(
    const float* __restrict__ H, int M,
    const float* __restrict__ gamma, const float* __restrict__ beta,
    float* __restrict__ scale, float* __restrict__ shift) {
  __shared__ float ssum[256], ssq[256];
  const int f = blockIdx.x;
  float s = 0.f, q = 0.f;
  for (int r = threadIdx.x; r < M; r += 256) {
    float v = H[r * FDIM + f];
    s += v;
    q += v * v;
  }
  ssum[threadIdx.x] = s;
  ssq[threadIdx.x] = q;
  __syncthreads();
  for (int off = 128; off > 0; off >>= 1) {
    if ((int)threadIdx.x < off) {
      ssum[threadIdx.x] += ssum[threadIdx.x + off];
      ssq[threadIdx.x]  += ssq[threadIdx.x + off];
    }
    __syncthreads();
  }
  if (threadIdx.x == 0) {
    float mean = ssum[0] / (float)M;
    float var  = fmaxf(ssq[0] / (float)M - mean * mean, 0.f);
    float sc   = gamma[f] * rsqrtf(var + BN_EPS);
    scale[f] = sc;
    shift[f] = beta[f] - mean * sc;
  }
}

// ---------------------------------------------------------------------------
// kNN (K=3): one wave32 per query; lane-local sorted top-3 + shfl_xor merge
// ---------------------------------------------------------------------------
__global__ __launch_bounds__(256) void knn_top3(
    const float* __restrict__ p1, const float* __restrict__ p2,
    int* __restrict__ idxOut, float* __restrict__ wOut) {
  const int q = blockIdx.x * 8 + (threadIdx.x >> 5);
  const int lane = threadIdx.x & 31;
  const int b = q >> 13;              // q / 8192
  const int base1 = b * NPTS1;
  const float qx = p2[q * 3 + 0];
  const float qy = p2[q * 3 + 1];
  const float qz = p2[q * 3 + 2];

  float d0 = 3.0e38f, d1 = 3.0e38f, d2 = 3.0e38f;
  int   i0 = -1, i1 = -1, i2 = -1;

  for (int i = lane; i < NPTS1; i += 32) {
    const int gi = base1 + i;
    const float dx = qx - p1[gi * 3 + 0];
    const float dy = qy - p1[gi * 3 + 1];
    const float dz = qz - p1[gi * 3 + 2];
    const float d = dx * dx + dy * dy + dz * dz;
    if (d < d2) {
      d2 = d; i2 = gi;
      if (d2 < d1) { float t = d1; d1 = d2; d2 = t; int u = i1; i1 = i2; i2 = u; }
      if (d1 < d0) { float t = d0; d0 = d1; d1 = t; int u = i0; i0 = i1; i1 = u; }
    }
  }
  #pragma unroll
  for (int off = 16; off >= 1; off >>= 1) {
    float ed[3];
    int   ej[3];
    ed[0] = __shfl_xor(d0, off, 32); ej[0] = __shfl_xor(i0, off, 32);
    ed[1] = __shfl_xor(d1, off, 32); ej[1] = __shfl_xor(i1, off, 32);
    ed[2] = __shfl_xor(d2, off, 32); ej[2] = __shfl_xor(i2, off, 32);
    #pragma unroll
    for (int t = 0; t < 3; ++t) {
      const float d = ed[t];
      const int  gi = ej[t];
      if (d < d2) {
        d2 = d; i2 = gi;
        if (d2 < d1) { float x = d1; d1 = d2; d2 = x; int u = i1; i1 = i2; i2 = u; }
        if (d1 < d0) { float x = d0; d0 = d1; d1 = x; int u = i0; i0 = i1; i1 = u; }
      }
    }
  }
  if (lane == 0) {
    const float w0 = 1.f / fmaxf(d0, 1e-16f);
    const float w1 = 1.f / fmaxf(d1, 1e-16f);
    const float w2 = 1.f / fmaxf(d2, 1e-16f);
    const float inv = 1.f / (w0 + w1 + w2);
    wOut[q * 3 + 0] = w0 * inv;
    wOut[q * 3 + 1] = w1 * inv;
    wOut[q * 3 + 2] = w2 * inv;
    idxOut[q * 3 + 0] = i0;
    idxOut[q * 3 + 1] = i1;
    idxOut[q * 3 + 2] = i2;
  }
}

// ---------------------------------------------------------------------------
// out = s1 * (sum_k w_k * H1[idx_k]) + t1 + s2 * H2 + t2
// ---------------------------------------------------------------------------
__global__ __launch_bounds__(256) void combine_out(
    const float* __restrict__ H1, const float* __restrict__ H2,
    const float* __restrict__ s1, const float* __restrict__ t1,
    const float* __restrict__ s2, const float* __restrict__ t2,
    const int* __restrict__ idx, const float* __restrict__ w,
    float* __restrict__ out) {
  const int q = blockIdx.x;
  const int f = threadIdx.x;
  const int i0 = idx[q * 3 + 0], i1 = idx[q * 3 + 1], i2 = idx[q * 3 + 2];
  const float w0 = w[q * 3 + 0], w1 = w[q * 3 + 1], w2 = w[q * 3 + 2];
  const float acc = w0 * H1[i0 * FDIM + f] +
                    w1 * H1[i1 * FDIM + f] +
                    w2 * H1[i2 * FDIM + f];
  out[q * FDIM + f] = s1[f] * acc + t1[f] + s2[f] * H2[q * FDIM + f] + t2[f];
}

// ---------------------------------------------------------------------------
// Pass-through tail outputs: positions_2 (f32), batch_2 (i64 -> f32)
// ---------------------------------------------------------------------------
__global__ void tail_copy(const float* __restrict__ pos2,
                          const long long* __restrict__ batch2,
                          float* __restrict__ out, int out_size) {
  const int tid = blockIdx.x * blockDim.x + threadIdx.x;
  const int featN = M2 * FDIM;       // 8388608
  const int posN = M2 * 3;           // 98304
  if (tid < posN) {
    const int o = featN + tid;
    if (o < out_size) out[o] = pos2[tid];
  } else if (tid < posN + M2) {
    const int t = tid - posN;
    const int o = featN + posN + t;
    if (o < out_size) out[o] = (float)batch2[t];
  }
}

// ---------------------------------------------------------------------------
extern "C" void kernel_launch(void* const* d_in, const int* in_sizes, int n_in,
                              void* d_out, int out_size, void* d_ws, size_t ws_size,
                              hipStream_t stream) {
  (void)in_sizes; (void)n_in; (void)ws_size;
  const float*     features_1  = (const float*)d_in[0];
  const float*     positions_1 = (const float*)d_in[1];
  const float*     features_2  = (const float*)d_in[3];
  const float*     positions_2 = (const float*)d_in[4];
  const long long* batch_2     = (const long long*)d_in[5];
  const float* W1 = (const float*)d_in[6];
  const float* b1 = (const float*)d_in[7];
  const float* g1 = (const float*)d_in[8];
  const float* be1 = (const float*)d_in[9];
  const float* W2 = (const float*)d_in[10];
  const float* b2 = (const float*)d_in[11];
  const float* g2 = (const float*)d_in[12];
  const float* be2 = (const float*)d_in[13];
  float* out = (float*)d_out;

  uint8_t* ws = (uint8_t*)d_ws;
  size_t off = 0;
  float* H1 = (float*)(ws + off); off += (size_t)M1 * FDIM * sizeof(float);
  float* H2 = (float*)(ws + off); off += (size_t)M2 * FDIM * sizeof(float);
  float* s1 = (float*)(ws + off); off += FDIM * sizeof(float);
  float* t1 = (float*)(ws + off); off += FDIM * sizeof(float);
  float* s2 = (float*)(ws + off); off += FDIM * sizeof(float);
  float* t2 = (float*)(ws + off); off += FDIM * sizeof(float);
  int*   idx = (int*)(ws + off);  off += (size_t)M2 * 3 * sizeof(int);
  float* wgt = (float*)(ws + off); off += (size_t)M2 * 3 * sizeof(float);

  const dim3 gblock(32, 4);  // 4 waves/block, one 16x16 tile per wave
  gemm_bias_wmma<<<dim3(FDIM / 64, M1 / 16), gblock, 0, stream>>>(features_1, W1, b1, H1);
  gemm_bias_wmma<<<dim3(FDIM / 64, M2 / 16), gblock, 0, stream>>>(features_2, W2, b2, H2);
  bn_stats<<<FDIM, 256, 0, stream>>>(H1, M1, g1, be1, s1, t1);
  bn_stats<<<FDIM, 256, 0, stream>>>(H2, M2, g2, be2, s2, t2);
  knn_top3<<<M2 / 8, 256, 0, stream>>>(positions_1, positions_2, idx, wgt);
  combine_out<<<M2, 256, 0, stream>>>(H1, H2, s1, t1, s2, t2, idx, wgt, out);
  const int tailN = M2 * 3 + M2;
  tail_copy<<<(tailN + 255) / 256, 256, 0, stream>>>(positions_2, batch_2, out, out_size);
}
